// DeeperHyperbolicEncoder_33801392620161
// MI455X (gfx1250) — compile-verified, matrix operand order checked by reading, not run on hardware
//
#include <hip/hip_runtime.h>

// ---------------------------------------------------------------------------
// Fused DeeperHyperbolicEncoder forward for MI455X (gfx1250, wave32, WMMA)
// GEMMs: bf16 WMMA, f32 accumulate. Weights staged into LDS by the Tensor
// Data Mover (double-buffered, padded for conflict-free ds_load_b128 reads).
// B-fragments and A-fragments software-pipelined so WMMA never waits a full
// LDS/L2 round trip.
// ---------------------------------------------------------------------------

typedef __attribute__((ext_vector_type(16))) __bf16 v16bf;
typedef __attribute__((ext_vector_type(8)))  __bf16 bf16x8;
typedef __attribute__((ext_vector_type(8)))  float  v8f;
typedef __attribute__((ext_vector_type(4)))  float  f32x4;
typedef __attribute__((ext_vector_type(4)))  unsigned int v4u;
typedef __attribute__((ext_vector_type(8)))  int          v8i;
typedef __attribute__((ext_vector_type(4)))  int          v4i;

#define HEPS   1e-15f
#define MAXN   0.996f            /* 1 - 4e-3 ball eps */
#define ATCLIP (1.0f - 1e-7f)

#define D_IN  512
#define D_H   256
#define D_OUT 128

// LDS layout (bytes). Staged weight rows: 64B data + 16B TDM pad = 80B stride
// (20 dwords -> 16 B-frag lanes hit disjoint bank groups).
#define W1TILE_B (256 * 80)      /* 20480 */
#define W2TILE_B (128 * 80)      /* 10240 */
#define OFF_W1_0 0
#define OFF_W1_1 (OFF_W1_0 + W1TILE_B)
#define OFF_W2_0 (OFF_W1_1 + W1TILE_B)
#define OFF_W2_1 (OFF_W2_0 + W2TILE_B)
#define OFF_U    (OFF_W2_1 + W2TILE_B)
#define U_STRIDE 528             /* 264 bf16: conflict-free A-frag reads     */
#define SMEM_B   (OFF_U + 128 * U_STRIDE)   /* 129,024 B */

__device__ __forceinline__ float rsum16(float v) {
  v += __shfl_xor(v, 1, 32);
  v += __shfl_xor(v, 2, 32);
  v += __shfl_xor(v, 4, 32);
  v += __shfl_xor(v, 8, 32);
  return v;
}

__device__ __forceinline__ float ftanh(float x) {
  x = fminf(fmaxf(x, -12.f), 12.f);
  float e = __expf(2.f * x);
  return (e - 1.f) / (e + 1.f);
}

__device__ __forceinline__ float fartanh(float x) {
  x = fminf(fmaxf(x, -ATCLIP), ATCLIP);
  return 0.5f * __logf((1.f + x) / (1.f - x));
}

// ---- Tensor Data Mover: 2D tile (32 x rows) of bf16, row-padded 64B+16B ---
// D# per CDNA5 ISA ch.8: group0 {count,lds,global,type}, group1 {flags,dims}.
__device__ __forceinline__ void tdm_load_tile(void* lds, const __bf16* gsrc,
                                              int rows, int rowlen) {
  unsigned lds_addr = (unsigned)(unsigned long long)lds;   // LDS aperture low32
  unsigned long long ga = (unsigned long long)gsrc;
  v4u g0;
  g0[0] = 1u;                                              // count=1, user mode
  g0[1] = lds_addr;
  g0[2] = (unsigned)(ga & 0xffffffffu);
  g0[3] = (unsigned)((ga >> 32) & 0x01ffffffu) | (2u << 30);  // addr[56:32]|type=2
  v8i g1;
  g1[0] = (int)((1u << 16)      // data_size = 2 bytes
              | (1u << 20)      // pad_enable
              | (3u << 22)      // pad_interval: 16 dwords (64B)
              | (3u << 25));    // pad_amount:   4 dwords (16B)
  g1[1] = (int)((unsigned)rowlen << 16);   // tensor_dim0[15:0]
  g1[2] = (int)((unsigned)rows   << 16);   // tensor_dim1[15:0]
  g1[3] = (int)(32u << 16);                // tile_dim0 = 32 (one K-slab)
  g1[4] = (int)(unsigned)rows;             // tile_dim1 = rows, tile_dim2 = 0
  g1[5] = (int)(unsigned)rowlen;           // tensor_dim0_stride lo
  g1[6] = 0;
  g1[7] = 0;
  v4i gz4 = {0, 0, 0, 0};                  // 2D tensor: groups 2/3 unused
  v8i gz8 = {0, 0, 0, 0, 0, 0, 0, 0};
  __builtin_amdgcn_tensor_load_to_lds(g0, g1, gz4, gz4, gz8, 0);
}

// --------- prologue: W1,W2 f32 -> bf16 (weights then live in L2) ------------
__global__ void cvt_weights_kernel(const float* __restrict__ W1,
                                   const float* __restrict__ W2,
                                   __bf16* __restrict__ w1b,
                                   __bf16* __restrict__ w2b) {
  int i = blockIdx.x * 256 + threadIdx.x;
  if (i < D_H * D_IN)  w1b[i] = (__bf16)W1[i];
  if (i < D_OUT * D_H) w2b[i] = (__bf16)W2[i];
}

// --------- fused forward ----------------------------------------------------
__global__ __launch_bounds__(256) void fused_hyp_kernel(
    const float*  __restrict__ x,
    const __bf16* __restrict__ W1b,
    const float*  __restrict__ b1,
    const __bf16* __restrict__ W2b,
    const float*  __restrict__ b2,
    float*        __restrict__ out) {

  __shared__ __align__(16) unsigned char smem[SMEM_B];

  const int lane = threadIdx.x & 31;
  const int wave = threadIdx.x >> 5;
  const int l15  = lane & 15;
  const int hi   = lane >> 4;             // 0: K {0..7,16..23}  1: K {8..15,24..31}
  const int strip = blockIdx.x * 128 + wave * 16;
  const bool tdmw = (wave == 0);          // one wave drives the TDM

  // ======================= GEMM1: mx = x @ W1.T (16 x 256) =================
  if (tdmw) {
    tdm_load_tile(smem + OFF_W1_0, W1b, 256, D_IN);
    __builtin_amdgcn_s_wait_tensorcnt(0);
  }
  __syncthreads();

  v8f acc[16];
#pragma unroll
  for (int t = 0; t < 16; ++t)
#pragma unroll
    for (int j = 0; j < 8; ++j) acc[t][j] = 0.f;

  const float* xrow = x + (size_t)(strip + l15) * D_IN;

  // A-fragment software pipeline: preload k-slab 0
  f32x4 c0 = *(const f32x4*)(xrow + hi * 8);
  f32x4 c1 = *(const f32x4*)(xrow + hi * 8 + 4);
  f32x4 c2 = *(const f32x4*)(xrow + hi * 8 + 16);
  f32x4 c3 = *(const f32x4*)(xrow + hi * 8 + 20);

  for (int kk = 0, t = 0; kk < D_IN; kk += 32, t ^= 1) {
    if (tdmw && (kk + 32) < D_IN)                       // DMA next weight tile
      tdm_load_tile(smem + (t ? OFF_W1_0 : OFF_W1_1), W1b + (kk + 32), 256, D_IN);

    // issue next A-slab loads now; consumed after the barrier next iteration
    const int kn = ((kk + 32) < D_IN ? (kk + 32) : 0) + hi * 8;
    f32x4 n0 = *(const f32x4*)(xrow + kn);
    f32x4 n1 = *(const f32x4*)(xrow + kn + 4);
    f32x4 n2 = *(const f32x4*)(xrow + kn + 16);
    f32x4 n3 = *(const f32x4*)(xrow + kn + 20);

    v16bf a;
#pragma unroll
    for (int j = 0; j < 4; ++j) {
      a[j]      = (__bf16)c0[j];
      a[4 + j]  = (__bf16)c1[j];
      a[8 + j]  = (__bf16)c2[j];
      a[12 + j] = (__bf16)c3[j];
    }

    const unsigned char* bbuf = smem + (t ? OFF_W1_1 : OFF_W1_0);

    // B-fragment two-slot rotation: load nt+1 while WMMA nt executes
    bf16x8 sA[2], sB[2];
    {
      const unsigned char* bp = bbuf + l15 * 80 + hi * 16;
      sA[0] = *(const bf16x8*)(bp);
      sB[0] = *(const bf16x8*)(bp + 32);
    }
#pragma unroll
    for (int nt = 0; nt < 16; ++nt) {
      const int cur = nt & 1, nxt = cur ^ 1;
      if (nt + 1 < 16) {
        const unsigned char* bp = bbuf + ((nt + 1) * 16 + l15) * 80 + hi * 16;
        sA[nxt] = *(const bf16x8*)(bp);
        sB[nxt] = *(const bf16x8*)(bp + 32);
      }
      v16bf bfr;
#pragma unroll
      for (int j = 0; j < 8; ++j) { bfr[j] = sA[cur][j]; bfr[8 + j] = sB[cur][j]; }
      acc[nt] = __builtin_amdgcn_wmma_f32_16x16x32_bf16(
          false, a, false, bfr, (short)0, acc[nt], false, false);
    }

    c0 = n0; c1 = n1; c2 = n2; c3 = n3;

    if (tdmw) __builtin_amdgcn_s_wait_tensorcnt(0);
    __syncthreads();
  }

  // W2 tile0 DMA overlaps the whole nonlinear stage below
  if (tdmw) tdm_load_tile(smem + OFF_W2_0, W2b, 128, D_H);

  // ============ per-row hyperbolic stage 1 ==================================
  float b1v[16];
#pragma unroll
  for (int nt = 0; nt < 16; ++nt) b1v[nt] = b1[nt * 16 + l15];
  float y2b = 0.f;
#pragma unroll
  for (int nt = 0; nt < 16; ++nt) y2b += b1v[nt] * b1v[nt];
  y2b = rsum16(y2b);

  float xnv[8];

#pragma unroll
  for (int v = 0; v < 8; ++v) {
    float s = 0.f;
#pragma unroll
    for (int nt = 0; nt < 16; ++nt) { float q = acc[nt][v]; s += q * q; }
    s = rsum16(s);
    float n1 = fmaxf(__builtin_sqrtf(s), HEPS);
    float t1 = ftanh(n1) / n1;                 // expmap0: h = t1 * mx
    float x2 = t1 * t1 * s;
    float xyp = 0.f;
#pragma unroll
    for (int nt = 0; nt < 16; ++nt) xyp += acc[nt][v] * b1v[nt];
    float xy  = t1 * rsum16(xyp);
    float ca  = 1.f + 2.f * xy + y2b;
    float cb  = 1.f - x2;
    float rdn = 1.f / fmaxf(1.f + 2.f * xy + x2 * y2b, HEPS);
    float s2 = 0.f;
#pragma unroll
    for (int nt = 0; nt < 16; ++nt) {
      float g = (ca * t1 * acc[nt][v] + cb * b1v[nt]) * rdn;
      acc[nt][v] = g;
      s2 += g * g;
    }
    s2 = rsum16(s2);
    float n2  = fmaxf(__builtin_sqrtf(s2), HEPS);
    float prj = (n2 > MAXN) ? (MAXN / n2) : 1.f;
    float n3  = fmaxf(n2 * prj, HEPS);
    float la  = fartanh(n3) / n3 * prj;        // project + logmap0 factor
    float s3 = 0.f;
#pragma unroll
    for (int nt = 0; nt < 16; ++nt) {
      float w = ftanh(la * acc[nt][v]);
      acc[nt][v] = w;
      s3 += w * w;
    }
    s3 = rsum16(s3);
    float nw = fmaxf(__builtin_sqrtf(s3), HEPS);
    float t2 = ftanh(nw) / nw;                 // u = t2*w, ||u|| = tanh(nw)
    xnv[v] = fmaxf(t2 * nw, HEPS);
    // C-layout -> A-layout transpose via this wave's private LDS rows
    __bf16* dst = (__bf16*)(smem + OFF_U + (size_t)(wave * 16 + v + hi * 8) * U_STRIDE);
#pragma unroll
    for (int nt = 0; nt < 16; ++nt)
      dst[nt * 16 + l15] = (__bf16)(t2 * acc[nt][v]);
  }

  // ======================= GEMM2: mx2 = u @ W2.T (16 x 128) ================
  if (tdmw) __builtin_amdgcn_s_wait_tensorcnt(0);
  __syncthreads();

  v8f acc2[8];
#pragma unroll
  for (int t = 0; t < 8; ++t)
#pragma unroll
    for (int j = 0; j < 8; ++j) acc2[t][j] = 0.f;

  const unsigned char* urow = smem + OFF_U + (size_t)(wave * 16 + l15) * U_STRIDE;

  for (int kk = 0, t = 0; kk < D_H; kk += 32, t ^= 1) {
    if (tdmw && (kk + 32) < D_H)
      tdm_load_tile(smem + (t ? OFF_W2_0 : OFF_W2_1), W2b + (kk + 32), 128, D_H);

    const unsigned char* bbuf = smem + (t ? OFF_W2_1 : OFF_W2_0);

    const int kb = kk + hi * 8;
    bf16x8 a0 = *(const bf16x8*)(urow + kb * 2);
    bf16x8 a1 = *(const bf16x8*)(urow + kb * 2 + 32);
    v16bf a;
#pragma unroll
    for (int j = 0; j < 8; ++j) { a[j] = a0[j]; a[8 + j] = a1[j]; }

    bf16x8 sA[2], sB[2];
    {
      const unsigned char* bp = bbuf + l15 * 80 + hi * 16;
      sA[0] = *(const bf16x8*)(bp);
      sB[0] = *(const bf16x8*)(bp + 32);
    }
#pragma unroll
    for (int nt = 0; nt < 8; ++nt) {
      const int cur = nt & 1, nxt = cur ^ 1;
      if (nt + 1 < 8) {
        const unsigned char* bp = bbuf + ((nt + 1) * 16 + l15) * 80 + hi * 16;
        sA[nxt] = *(const bf16x8*)(bp);
        sB[nxt] = *(const bf16x8*)(bp + 32);
      }
      v16bf bfr;
#pragma unroll
      for (int j = 0; j < 8; ++j) { bfr[j] = sA[cur][j]; bfr[8 + j] = sB[cur][j]; }
      acc2[nt] = __builtin_amdgcn_wmma_f32_16x16x32_bf16(
          false, a, false, bfr, (short)0, acc2[nt], false, false);
    }

    if (tdmw) __builtin_amdgcn_s_wait_tensorcnt(0);
    __syncthreads();
  }

  // ============ per-row hyperbolic stage 2 + store ==========================
  float b2v[8];
#pragma unroll
  for (int nt = 0; nt < 8; ++nt) b2v[nt] = b2[nt * 16 + l15];
  float y2c = 0.f;
#pragma unroll
  for (int nt = 0; nt < 8; ++nt) y2c += b2v[nt] * b2v[nt];
  y2c = rsum16(y2c);

#pragma unroll
  for (int v = 0; v < 8; ++v) {
    float s = 0.f;
#pragma unroll
    for (int nt = 0; nt < 8; ++nt) { float q = acc2[nt][v]; s += q * q; }
    s = rsum16(s);
    float xn  = xnv[v];
    float mxn = fmaxf(__builtin_sqrtf(s), HEPS);
    float rf  = ftanh(mxn / xn * fartanh(xn)) / mxn;   // mobius_matvec factor
    rf        = (s > 0.f) ? rf : 0.f;                  // zero-row guard
    float x2  = rf * rf * s;
    float xyp = 0.f;
#pragma unroll
    for (int nt = 0; nt < 8; ++nt) xyp += acc2[nt][v] * b2v[nt];
    float xy  = rf * rsum16(xyp);
    float ca  = 1.f + 2.f * xy + y2c;
    float cb  = 1.f - x2;
    float rdn = 1.f / fmaxf(1.f + 2.f * xy + x2 * y2c, HEPS);
    float s2 = 0.f;
#pragma unroll
    for (int nt = 0; nt < 8; ++nt) {
      float g = (ca * rf * acc2[nt][v] + cb * b2v[nt]) * rdn;
      acc2[nt][v] = g;
      s2 += g * g;
    }
    s2 = rsum16(s2);
    float n2  = fmaxf(__builtin_sqrtf(s2), HEPS);
    float prj = (n2 > MAXN) ? (MAXN / n2) : 1.f;       // project (idempotent)
    float* orow = out + (size_t)(strip + v + hi * 8) * D_OUT;
#pragma unroll
    for (int nt = 0; nt < 8; ++nt)
      orow[nt * 16 + l15] = acc2[nt][v] * prj;
  }
}

// --------------------------------------------------------------------------
extern "C" void kernel_launch(void* const* d_in, const int* in_sizes, int n_in,
                              void* d_out, int out_size, void* d_ws, size_t ws_size,
                              hipStream_t stream) {
  const float* x  = (const float*)d_in[0];   // (131072, 512)
  const float* W1 = (const float*)d_in[1];   // (256, 512)
  const float* b1 = (const float*)d_in[2];   // (256,)
  const float* W2 = (const float*)d_in[3];   // (128, 256)
  const float* b2 = (const float*)d_in[4];   // (128,)
  float* out = (float*)d_out;                // (131072, 128)

  __bf16* w1b = (__bf16*)d_ws;               // 256*512 bf16
  __bf16* w2b = w1b + D_H * D_IN;            // 128*256 bf16

  cvt_weights_kernel<<<dim3(512), dim3(256), 0, stream>>>(W1, W2, w1b, w2b);

  const int rows = in_sizes[0] / D_IN;       // 131072
  fused_hyp_kernel<<<dim3(rows / 128), dim3(256), 0, stream>>>(
      x, w1b, b1, w2b, b2, out);
}
